// MultiBranchAttention_27977416966383
// MI455X (gfx1250) — matrix operand order repaired and checked by reading, not compiled, not run on hardware
//
#include <hip/hip_runtime.h>

#define C_DIM 64
#define CR    4
#define B_SZ  2
#define HW    6400   // 80*80
#define KB    64     // keys per inner-loop chunk

typedef __attribute__((ext_vector_type(2)))  float    v2f;
typedef __attribute__((ext_vector_type(8)))  float    v8f;
typedef __attribute__((ext_vector_type(8)))  _Float16 v8h;
typedef __attribute__((ext_vector_type(16))) _Float16 v16h;

// ---------------------------------------------------------------------------
// Kernel 1: fused 1x1-conv projections.
//   qab/kab/qbc/kbc: f32 [B][N][4]   (position-major, channel-contiguous)
//   vt:              f16 [B][64][N]  (channel-major -> contiguous key runs)
// block = (64 couts, 4 positions); grid = (N/4, B)
// ---------------------------------------------------------------------------
__global__ __launch_bounds__(256) void mba_proj_kernel(
    const float* __restrict__ a, const float* __restrict__ b, const float* __restrict__ c,
    const float* __restrict__ wq_ab, const float* __restrict__ bq_ab,
    const float* __restrict__ wk_ab, const float* __restrict__ bk_ab,
    const float* __restrict__ wq_bc, const float* __restrict__ bq_bc,
    const float* __restrict__ wk_bc, const float* __restrict__ bk_bc,
    const float* __restrict__ wv,    const float* __restrict__ bv,
    float* __restrict__ qab, float* __restrict__ kab,
    float* __restrict__ qbc, float* __restrict__ kbc,
    _Float16* __restrict__ vt)
{
  const int cout = threadIdx.x;                       // 0..63
  const int n    = blockIdx.x * 4 + threadIdx.y;      // spatial position
  const int bz   = blockIdx.y;

  const float* bsrc = b + (size_t)bz * C_DIM * HW + n;

  // value projection: v[cout] = wv[cout,:] . b[:,n] + bv[cout]
  float accv = bv[cout];
  #pragma unroll 8
  for (int cc = 0; cc < C_DIM; ++cc)
    accv = fmaf(wv[cout * C_DIM + cc], bsrc[(size_t)cc * HW], accv);
  vt[((size_t)bz * C_DIM + cout) * HW + n] = (_Float16)accv;

  // q/k projections: 16 lanes cover 4 outputs x 4 channels
  if (cout < 16) {
    const int d     = cout & 3;
    const int which = cout >> 2;     // 0 qab, 1 kab, 2 qbc, 3 kbc
    const float *w, *bias, *x; float* dst;
    switch (which) {
      case 0:  w = wq_ab; bias = bq_ab; x = a; dst = qab; break;
      case 1:  w = wk_ab; bias = bk_ab; x = b; dst = kab; break;
      case 2:  w = wq_bc; bias = bq_bc; x = c; dst = qbc; break;
      default: w = wk_bc; bias = bk_bc; x = b; dst = kbc; break;
    }
    const float* xs = x + (size_t)bz * C_DIM * HW + n;
    float acc = bias[d];
    #pragma unroll 8
    for (int cc = 0; cc < C_DIM; ++cc)
      acc = fmaf(w[d * C_DIM + cc], xs[(size_t)cc * HW], acc);
    dst[((size_t)bz * HW + n) * CR + d] = acc;
  }
}

// ---------------------------------------------------------------------------
// Kernel 2: flash attention, one wave per 16-query tile, 64 keys per step.
//   scores : 4x V_WMMA_F32_16X16X4_F32  (CR=4 exact fit, fp32)
//   P.V    : 8x V_WMMA_F32_16X16X32_F16 (f32 accumulate)
// Softmax is done row-serially: each lane owns one (row, 32-col half) of the
// score tile -> registers only, 2 ds_bpermute per chunk total.
// ---------------------------------------------------------------------------
__global__ __launch_bounds__(32) void mba_flash_kernel(
    const float* __restrict__ qab, const float* __restrict__ kab,
    const float* __restrict__ qbc, const float* __restrict__ kbc,
    const _Float16* __restrict__ vt,
    float* __restrict__ tmp)
{
  __shared__ __align__(16) float    lds_s[16 * KB];   // score tile, 4 KB
  __shared__ __align__(16) _Float16 lds_p[16 * KB];   // prob tile,  2 KB
  __shared__ __align__(16) float    lds_bc[16];       // per-row broadcast

  const int lane = threadIdx.x;
  const int g    = lane >> 4;      // half-wave group (WMMA layout)
  const int ln   = lane & 15;
  const int rowId = lane >> 1;     // softmax: row owned by this lane
  const int h     = lane & 1;      // softmax: which 32-col half
  const int q0   = blockIdx.x * 16;
  const int branch = blockIdx.y;
  const int bz   = blockIdx.z;

  const float* q = (branch == 0) ? qab : qbc;
  const float* k = (branch == 0) ? kab : kbc;
  float* outp = tmp + ((size_t)branch * B_SZ + bz) * C_DIM * HW;

  const float*    kbase  = k  + (size_t)bz * HW * CR;
  const _Float16* vtbase = vt + (size_t)bz * C_DIM * HW;

  // Q A-fragment (16x4 f32): lane (g,ln) holds (m=ln, k=2g,2g+1) -> b64 load
  const v2f aq = *(const v2f*)&q[((size_t)bz * HW + q0 + ln) * CR + 2 * g];

  float mrun = -3.0e38f;           // running max of owned row
  float lrun = 0.0f;               // running sum of owned row
  const v8f vzero = {};
  v8f acc[4];
  #pragma unroll
  for (int t = 0; t < 4; ++t) acc[t] = vzero;

  for (int kb = 0; kb < HW; kb += KB) {
    // ---- scores: 4 tiles of 16 keys, exact-fit f32 WMMA ----
    v8f s[4];
    #pragma unroll
    for (int c = 0; c < 4; ++c) {
      const v2f bk = *(const v2f*)&kbase[(size_t)(kb + 16 * c + ln) * CR + 2 * g];
      s[c] = __builtin_amdgcn_wmma_f32_16x16x4_f32(false, aq, false, bk,
                                                   (short)0, vzero, false, false);
    }
    if (kb + KB < HW)
      __builtin_prefetch(vtbase + (size_t)ln * HW + kb + KB, 0, 0);

    // ---- dump score tile to LDS in row-major (C/D layout stores) ----
    #pragma unroll
    for (int c = 0; c < 4; ++c)
      #pragma unroll
      for (int r = 0; r < 8; ++r)
        lds_s[(8 * g + r) * KB + 16 * c + ln] = s[c][r];
    __syncthreads();

    // ---- row-serial softmax: lane owns (rowId, cols h*32..h*32+31) ----
    float vals[32];
    {
      const float4* srow = (const float4*)&lds_s[rowId * KB + h * 32];
      #pragma unroll
      for (int i = 0; i < 8; ++i) {
        const float4 t4 = srow[i];
        vals[4 * i + 0] = t4.x; vals[4 * i + 1] = t4.y;
        vals[4 * i + 2] = t4.z; vals[4 * i + 3] = t4.w;
      }
    }
    float cmax = vals[0];
    #pragma unroll
    for (int i = 1; i < 32; ++i) cmax = fmaxf(cmax, vals[i]);
    cmax = fmaxf(cmax, __shfl_xor(cmax, 1, 32));        // combine halves

    const float mn   = fmaxf(mrun, cmax);
    const float corr = __expf(mrun - mn);
    mrun = mn;

    float rsum = 0.0f;
    v8h pvec[4];
    #pragma unroll
    for (int i = 0; i < 32; ++i) {
      const float e = __expf(vals[i] - mn);
      rsum += e;
      pvec[i >> 3][i & 7] = (_Float16)e;
    }
    rsum += __shfl_xor(rsum, 1, 32);                    // combine halves
    lrun = lrun * corr + rsum;

    // write P row-half (32 halves = 4x b128) and broadcast corr
    {
      v8h* prow = (v8h*)&lds_p[rowId * KB + h * 32];
      #pragma unroll
      for (int i = 0; i < 4; ++i) prow[i] = pvec[i];
    }
    if (h == 0) lds_bc[rowId] = corr;
    __syncthreads();

    // ---- rescale accumulators with per-row corr (C/D layout) ----
    {
      const float4* cb = (const float4*)&lds_bc[8 * g];
      const float4 c0 = cb[0], c1 = cb[1];
      const float cf[8] = {c0.x, c0.y, c0.z, c0.w, c1.x, c1.y, c1.z, c1.w};
      #pragma unroll
      for (int t = 0; t < 4; ++t)
        #pragma unroll
        for (int r = 0; r < 8; ++r)
          acc[t][r] *= cf[r];
    }

    // ---- P.V: 2 key-halves x 4 channel tiles of f16 WMMA ----
    #pragma unroll
    for (int c2 = 0; c2 < 2; ++c2) {
      // A-fragment of P (16x32 f16): two contiguous 8-half runs per lane
      const v8h pa_lo = *(const v8h*)&lds_p[ln * KB + c2 * 32 + 8 * g];
      const v8h pa_hi = *(const v8h*)&lds_p[ln * KB + c2 * 32 + 16 + 8 * g];
      const v16h pa = __builtin_shufflevector(pa_lo, pa_hi,
          0, 1, 2, 3, 4, 5, 6, 7, 8, 9, 10, 11, 12, 13, 14, 15);
      #pragma unroll
      for (int t = 0; t < 4; ++t) {
        // B-fragment of V (32x16 f16): ch = 16t+ln, two contiguous 8-position runs
        const _Float16* vp = vtbase + (size_t)(16 * t + ln) * HW
                                    + kb + c2 * 32 + 8 * g;
        const v8h vb_lo = *(const v8h*)vp;
        const v8h vb_hi = *(const v8h*)(vp + 16);
        const v16h vb = __builtin_shufflevector(vb_lo, vb_hi,
            0, 1, 2, 3, 4, 5, 6, 7, 8, 9, 10, 11, 12, 13, 14, 15);
        acc[t] = __builtin_amdgcn_wmma_f32_16x16x32_f16(false, pa, false, vb,
                                                        (short)0, acc[t], false, false);
      }
    }
    __syncthreads();   // protect lds_s / lds_p / lds_bc before next chunk
  }

  // ---- epilogue: broadcast row sums, normalize, store ----
  if (h == 0) lds_bc[rowId] = lrun;
  __syncthreads();
  {
    const float4* cb = (const float4*)&lds_bc[8 * g];
    const float4 l0 = cb[0], l1 = cb[1];
    const float linv[8] = {1.0f / l0.x, 1.0f / l0.y, 1.0f / l0.z, 1.0f / l0.w,
                           1.0f / l1.x, 1.0f / l1.y, 1.0f / l1.z, 1.0f / l1.w};
    #pragma unroll
    for (int t = 0; t < 4; ++t) {
      const int ch = 16 * t + ln;
      #pragma unroll
      for (int r = 0; r < 8; ++r)
        outp[(size_t)ch * HW + q0 + 8 * g + r] = acc[t][r] * linv[r];
    }
  }
}

// ---------------------------------------------------------------------------
// Kernel 3: out = 0.5 * (branch0 + branch1), float4-vectorized
// ---------------------------------------------------------------------------
__global__ __launch_bounds__(256) void mba_combine_kernel(
    const float4* __restrict__ t0, const float4* __restrict__ t1,
    float4* __restrict__ out, int n4)
{
  const int i = blockIdx.x * blockDim.x + threadIdx.x;
  if (i < n4) {
    const float4 x = t0[i], y = t1[i];
    float4 r;
    r.x = 0.5f * (x.x + y.x); r.y = 0.5f * (x.y + y.y);
    r.z = 0.5f * (x.z + y.z); r.w = 0.5f * (x.w + y.w);
    out[i] = r;
  }
}

// ---------------------------------------------------------------------------
extern "C" void kernel_launch(void* const* d_in, const int* in_sizes, int n_in,
                              void* d_out, int out_size, void* d_ws, size_t ws_size,
                              hipStream_t stream)
{
  (void)in_sizes; (void)n_in; (void)out_size; (void)ws_size;

  const float* a     = (const float*)d_in[0];
  const float* b     = (const float*)d_in[1];
  const float* c     = (const float*)d_in[2];
  const float* wq_ab = (const float*)d_in[3];
  const float* bq_ab = (const float*)d_in[4];
  const float* wk_ab = (const float*)d_in[5];
  const float* bk_ab = (const float*)d_in[6];
  const float* wq_bc = (const float*)d_in[7];
  const float* bq_bc = (const float*)d_in[8];
  const float* wk_bc = (const float*)d_in[9];
  const float* bk_bc = (const float*)d_in[10];
  const float* wv    = (const float*)d_in[11];
  const float* bv    = (const float*)d_in[12];
  float* out = (float*)d_out;

  char* ws = (char*)d_ws;
  size_t off = 0;
  auto alloc = [&](size_t bytes) -> char* {
    char* p = ws + off;
    off = (off + bytes + 255) & ~(size_t)255;
    return p;
  };
  float*    qab = (float*)   alloc((size_t)B_SZ * HW * CR * sizeof(float));
  float*    kab = (float*)   alloc((size_t)B_SZ * HW * CR * sizeof(float));
  float*    qbc = (float*)   alloc((size_t)B_SZ * HW * CR * sizeof(float));
  float*    kbc = (float*)   alloc((size_t)B_SZ * HW * CR * sizeof(float));
  _Float16* vt  = (_Float16*)alloc((size_t)B_SZ * C_DIM * HW * sizeof(_Float16));
  float*    tmp = (float*)   alloc((size_t)2 * B_SZ * C_DIM * HW * sizeof(float));

  // 1) projections
  mba_proj_kernel<<<dim3(HW / 4, B_SZ), dim3(64, 4), 0, stream>>>(
      a, b, c, wq_ab, bq_ab, wk_ab, bk_ab, wq_bc, bq_bc, wk_bc, bk_bc,
      wv, bv, qab, kab, qbc, kbc, vt);

  // 2) fused flash attention per branch
  mba_flash_kernel<<<dim3(HW / 16, 2, B_SZ), dim3(32), 0, stream>>>(
      qab, kab, qbc, kbc, vt, tmp);

  // 3) combine branches
  const int n4 = (B_SZ * C_DIM * HW) / 4;
  const float4* t0 = (const float4*)tmp;
  const float4* t1 = (const float4*)(tmp + (size_t)B_SZ * C_DIM * HW);
  mba_combine_kernel<<<dim3((n4 + 255) / 256), dim3(256), 0, stream>>>(
      t0, t1, (float4*)out, n4);
}
